// GroupedExperts_42949673289
// MI455X (gfx1250) — matrix operand order repaired and checked by reading, not compiled, not run on hardware
//
#include <hip/hip_runtime.h>
#include <hip/hip_bf16.h>
#include <stdint.h>

// MoE grouped experts on gfx1250 (wave32): bf16 WMMA + TDM tensor_load_to_lds.
// Pass 0: convert x -> bf16; convert+transpose weights -> bf16 [n][k] in d_ws.
// Pass 1: h = silu(x@gate) * (x@up)  (WMMA, TDM double-buffered)  -> bf16 ws
// Pass 2: out = h @ down             (WMMA, TDM double-buffered)  -> f32 out

#define NEXP   8
#define DIM    2048
#define HIDDEN 5632
#define LDA    40               // padded LDS row (bf16 elems): 80 B pitch
#define TILEB  (128 * LDA * 2)  // one padded 128x32 bf16 tile in LDS: 10240 B

typedef __attribute__((ext_vector_type(16))) __bf16 v16bf;
typedef __attribute__((ext_vector_type(8)))  float  v8f;
typedef __attribute__((ext_vector_type(4)))  unsigned int u32x4;
typedef __attribute__((ext_vector_type(8)))  int          i32x8;
typedef __attribute__((ext_vector_type(4)))  int          i32x4;

union FragBF { uint4 q[2]; v16bf v; };

__device__ __forceinline__ unsigned short f2bf(float f) {
  unsigned int u = __float_as_uint(f);
  u += 0x7FFFu + ((u >> 16) & 1u);          // RNE
  return (unsigned short)(u >> 16);
}

// ---- Tensor Data Mover: load one 128-row x 32-elem bf16 tile into LDS ----
// D# pad fields reproduce the LDA=40 pitch: rows are 16 DWORDs, pad_interval=3
// (every 16 DWORDs) inserts pad_amount=3 (4 DWORDs = 16 B) -> 80 B row pitch.
__device__ __forceinline__ void tdm_load_tile(unsigned lds_off, const void* gptr,
                                              unsigned stride_elems) {
  const uint64_t ga = (uint64_t)gptr;
  u32x4 g0;
  g0[0] = 1u;                                              // count=1, user mode
  g0[1] = lds_off;                                         // lds_addr (bytes)
  g0[2] = (unsigned)ga;                                    // global_addr[31:0]
  g0[3] = (unsigned)((ga >> 32) & 0x1FFFFFFu) | (2u << 30); // [56:32] | type=2
  i32x8 g1;
  g1[0] = (int)((1u << 16)   |   // data_size = 1 (2 bytes)
                (1u << 20)   |   // pad_enable
                (3u << 22)   |   // pad_interval = 16 DWORDs
                (3u << 25));     // pad_amount  = 4 DWORDs (16 B)
  g1[1] = (int)(32u << 16);                 // tensor_dim0 = 32 (lo16)
  g1[2] = (int)(128u << 16);                // tensor_dim1 = 128 (lo16)
  g1[3] = (int)(32u << 16);                 // tile_dim0 = 32
  g1[4] = (int)(128u);                      // tile_dim1 = 128, tile_dim2 = 0
  g1[5] = (int)stride_elems;                // tensor_dim0_stride lo32
  g1[6] = 0;                                // stride0 hi16 | stride1 lo16
  g1[7] = 0;                                // stride1 hi32
  i32x4 z4 = {0, 0, 0, 0};
#if __clang_major__ >= 23
  i32x8 z8 = {0, 0, 0, 0, 0, 0, 0, 0};
  __builtin_amdgcn_tensor_load_to_lds(g0, g1, z4, z4, z8, 0);
#else
  __builtin_amdgcn_tensor_load_to_lds(g0, g1, z4, z4, 0);
#endif
}

// ------------------------------------------------------------ converters ---
__global__ __launch_bounds__(256)
void cvt_x(const float* __restrict__ x, unsigned short* __restrict__ xb, int n4) {
  const int i = blockIdx.x * 256 + threadIdx.x;   // 4 floats per thread
  if (i < n4) {
    float4 f = ((const float4*)x)[i];
    uint2 t;
    t.x = (unsigned)f2bf(f.x) | ((unsigned)f2bf(f.y) << 16);
    t.y = (unsigned)f2bf(f.z) | ((unsigned)f2bf(f.w) << 16);
    ((uint2*)xb)[i] = t;
  }
}

// src: f32 [e][K][N]  ->  dst: bf16 [e][N][K]   (32x32 tiles through LDS)
__global__ __launch_bounds__(256)
void transconv(const float* __restrict__ src, unsigned short* __restrict__ dst,
               int K, int N) {
  __shared__ unsigned short t[32][36];
  const int e = blockIdx.z;
  src += (size_t)e * K * N;
  dst += (size_t)e * K * N;
  const int n0 = blockIdx.x * 32;
  const int k0 = blockIdx.y * 32;
  const int tx = threadIdx.x & 31;
  const int ty = threadIdx.x >> 5;               // 0..7
  #pragma unroll
  for (int i = 0; i < 4; ++i) {
    const int k = ty + i * 8;
    t[tx][k] = f2bf(src[(size_t)(k0 + k) * N + n0 + tx]);
  }
  __syncthreads();
  const int r = threadIdx.x >> 3;                // 0..31  (n within tile)
  const int c = (threadIdx.x & 7) * 4;           // 0..28  (k within tile)
  uint2 o;
  o.x = (unsigned)t[r][c]     | ((unsigned)t[r][c + 1] << 16);
  o.y = (unsigned)t[r][c + 2] | ((unsigned)t[r][c + 3] << 16);
  *(uint2*)&dst[(size_t)(n0 + r) * K + k0 + c] = o;
}

// ---------------------------------------------------------------- phase 1 ---
__global__ __launch_bounds__(256)
void moe_gateup(const unsigned short* __restrict__ xb,
                const unsigned short* __restrict__ gb,   // [e][HIDDEN][DIM]
                const unsigned short* __restrict__ ub,   // [e][HIDDEN][DIM]
                const int* __restrict__ counts,
                unsigned short* __restrict__ hbuf) {
  __shared__ __align__(16) unsigned short smem[2][3][128 * LDA]; // A,Bg,Bu x2

  const int e = blockIdx.z;
  int off = 0;
  for (int i = 0; i < e; ++i) off += counts[i];
  const int cnt = counts[e];

  const int bm = blockIdx.x * 128;
  const int bn = blockIdx.y * 128;
  if (bm >= cnt) return;

  const int tid  = threadIdx.x;
  const int lane = tid & 31;
  const int wid  = tid >> 5;
  const int wm   = wid & 1;
  const int wn   = wid >> 1;
  const int lr   = lane & 15;
  const int lh   = lane >> 4;

  const unsigned short* aT = xb + (size_t)(off + bm) * DIM;
  const unsigned short* gT = gb + (size_t)e * HIDDEN * DIM + (size_t)bn * DIM;
  const unsigned short* uT = ub + (size_t)e * HIDDEN * DIM + (size_t)bn * DIM;
  const unsigned smem_base = (unsigned)(uintptr_t)&smem[0][0][0];

  v8f accG[4][2], accU[4][2];
  const v8f vz = {0.f,0.f,0.f,0.f,0.f,0.f,0.f,0.f};
  for (int i = 0; i < 4; ++i)
    for (int j = 0; j < 2; ++j) { accG[i][j] = vz; accU[i][j] = vz; }

  // wave 0 issues 3 TDM tile loads for K-tile kt into buffer buf
  auto stage = [&](int buf, int kt) {
    const int k0 = kt * 32;
    const unsigned base = smem_base + (unsigned)buf * (3 * TILEB);
    tdm_load_tile(base,             aT + k0, DIM);
    tdm_load_tile(base + TILEB,     gT + k0, DIM);
    tdm_load_tile(base + 2 * TILEB, uT + k0, DIM);
  };

  const int KT = DIM / 32;
  if (wid == 0) stage(0, 0);
  for (int kt = 0; kt < KT; ++kt) {
    const int buf = kt & 1;
    if (wid == 0) {
      if (kt + 1 < KT) { stage(buf ^ 1, kt + 1); __builtin_amdgcn_s_wait_tensorcnt(3); }
      else             { __builtin_amdgcn_s_wait_tensorcnt(0); }
    }
    __syncthreads();

    const unsigned short* As = smem[buf][0];
    const unsigned short* Bg = smem[buf][1];
    const unsigned short* Bu = smem[buf][2];

    FragBF a[4], fg[2], fu[2];
    #pragma unroll
    for (int mi = 0; mi < 4; ++mi) {
      const int row = wm * 64 + mi * 16 + lr;
      a[mi].q[0] = *(const uint4*)&As[row * LDA + lh * 8];
      a[mi].q[1] = *(const uint4*)&As[row * LDA + 16 + lh * 8];
    }
    #pragma unroll
    for (int ni = 0; ni < 2; ++ni) {
      const int row = wn * 32 + ni * 16 + lr;
      fg[ni].q[0] = *(const uint4*)&Bg[row * LDA + lh * 8];
      fg[ni].q[1] = *(const uint4*)&Bg[row * LDA + 16 + lh * 8];
      fu[ni].q[0] = *(const uint4*)&Bu[row * LDA + lh * 8];
      fu[ni].q[1] = *(const uint4*)&Bu[row * LDA + 16 + lh * 8];
    }
    #pragma unroll
    for (int mi = 0; mi < 4; ++mi)
      #pragma unroll
      for (int ni = 0; ni < 2; ++ni) {
        accG[mi][ni] = __builtin_amdgcn_wmma_f32_16x16x32_bf16(
            false, a[mi].v, false, fg[ni].v, (short)0, accG[mi][ni], false, false);
        accU[mi][ni] = __builtin_amdgcn_wmma_f32_16x16x32_bf16(
            false, a[mi].v, false, fu[ni].v, (short)0, accU[mi][ni], false, false);
      }
    __syncthreads();  // all waves done reading before buf is overwritten
  }

  #pragma unroll
  for (int mi = 0; mi < 4; ++mi)
    #pragma unroll
    for (int ni = 0; ni < 2; ++ni)
      #pragma unroll
      for (int r = 0; r < 8; ++r) {
        const int m = wm * 64 + mi * 16 + lh * 8 + r;
        if (bm + m < cnt) {
          const int n = bn + wn * 32 + ni * 16 + lr;
          float g = accG[mi][ni][r];
          float u = accU[mi][ni][r];
          float s = g / (1.0f + __expf(-g));
          hbuf[(size_t)(off + bm + m) * HIDDEN + n] = f2bf(s * u);
        }
      }
}

// ---------------------------------------------------------------- phase 2 ---
__global__ __launch_bounds__(256)
void moe_down(const unsigned short* __restrict__ hbuf,
              const unsigned short* __restrict__ db,    // [e][DIM][HIDDEN]
              const int* __restrict__ counts,
              float* __restrict__ out) {
  __shared__ __align__(16) unsigned short smem[2][2][128 * LDA]; // A,B x2

  const int e = blockIdx.z;
  int off = 0;
  for (int i = 0; i < e; ++i) off += counts[i];
  const int cnt = counts[e];

  const int bm = blockIdx.x * 128;
  const int bn = blockIdx.y * 128;
  if (bm >= cnt) return;

  const int tid  = threadIdx.x;
  const int lane = tid & 31;
  const int wid  = tid >> 5;
  const int wm   = wid & 1;
  const int wn   = wid >> 1;
  const int lr   = lane & 15;
  const int lh   = lane >> 4;

  const unsigned short* aT = hbuf + (size_t)(off + bm) * HIDDEN;
  const unsigned short* bT = db + (size_t)e * DIM * HIDDEN + (size_t)bn * HIDDEN;
  const unsigned smem_base = (unsigned)(uintptr_t)&smem[0][0][0];

  v8f acc[4][2];
  const v8f vz = {0.f,0.f,0.f,0.f,0.f,0.f,0.f,0.f};
  for (int i = 0; i < 4; ++i)
    for (int j = 0; j < 2; ++j) acc[i][j] = vz;

  auto stage = [&](int buf, int kt) {
    const int k0 = kt * 32;
    const unsigned base = smem_base + (unsigned)buf * (2 * TILEB);
    tdm_load_tile(base,         aT + k0, HIDDEN);
    tdm_load_tile(base + TILEB, bT + k0, HIDDEN);
  };

  const int KT = HIDDEN / 32;
  if (wid == 0) stage(0, 0);
  for (int kt = 0; kt < KT; ++kt) {
    const int buf = kt & 1;
    if (wid == 0) {
      if (kt + 1 < KT) { stage(buf ^ 1, kt + 1); __builtin_amdgcn_s_wait_tensorcnt(2); }
      else             { __builtin_amdgcn_s_wait_tensorcnt(0); }
    }
    __syncthreads();

    const unsigned short* As = smem[buf][0];
    const unsigned short* Bs = smem[buf][1];

    FragBF a[4], b[2];
    #pragma unroll
    for (int mi = 0; mi < 4; ++mi) {
      const int row = wm * 64 + mi * 16 + lr;
      a[mi].q[0] = *(const uint4*)&As[row * LDA + lh * 8];
      a[mi].q[1] = *(const uint4*)&As[row * LDA + 16 + lh * 8];
    }
    #pragma unroll
    for (int ni = 0; ni < 2; ++ni) {
      const int row = wn * 32 + ni * 16 + lr;
      b[ni].q[0] = *(const uint4*)&Bs[row * LDA + lh * 8];
      b[ni].q[1] = *(const uint4*)&Bs[row * LDA + 16 + lh * 8];
    }
    #pragma unroll
    for (int mi = 0; mi < 4; ++mi)
      #pragma unroll
      for (int ni = 0; ni < 2; ++ni)
        acc[mi][ni] = __builtin_amdgcn_wmma_f32_16x16x32_bf16(
            false, a[mi].v, false, b[ni].v, (short)0, acc[mi][ni], false, false);
    __syncthreads();
  }

  #pragma unroll
  for (int mi = 0; mi < 4; ++mi)
    #pragma unroll
    for (int ni = 0; ni < 2; ++ni)
      #pragma unroll
      for (int r = 0; r < 8; ++r) {
        const int m = wm * 64 + mi * 16 + lh * 8 + r;
        if (bm + m < cnt) {
          const int n = bn + wn * 32 + ni * 16 + lr;
          out[(size_t)(off + bm + m) * DIM + n] = acc[mi][ni][r];
        }
      }
}

// ----------------------------------------------------------------- launch ---
extern "C" void kernel_launch(void* const* d_in, const int* in_sizes, int n_in,
                              void* d_out, int out_size, void* d_ws, size_t ws_size,
                              hipStream_t stream) {
  const float* x    = (const float*)d_in[0];
  const float* gate = (const float*)d_in[1];
  const float* up   = (const float*)d_in[2];
  const float* down = (const float*)d_in[3];
  const int*   cnts = (const int*)d_in[4];
  float* out = (float*)d_out;

  const int n_tokens = in_sizes[0] / DIM;                // 16384
  const int per_exp  = n_tokens / NEXP;                  // 2048
  const int mblocks  = (per_exp + 127) / 128;            // 16
  const size_t wsz   = (size_t)NEXP * DIM * HIDDEN;      // per weight tensor

  unsigned short* xb = (unsigned short*)d_ws;            // [nt][DIM]
  unsigned short* gb = xb + (size_t)n_tokens * DIM;      // [e][HIDDEN][DIM]
  unsigned short* ub = gb + wsz;                         // [e][HIDDEN][DIM]
  unsigned short* db = ub + wsz;                         // [e][DIM][HIDDEN]
  unsigned short* hb = db + wsz;                         // [nt][HIDDEN]

  dim3 blk(256, 1, 1);

  const int n4 = n_tokens * DIM / 4;
  cvt_x<<<dim3((n4 + 255) / 256, 1, 1), blk, 0, stream>>>(x, xb, n4);

  transconv<<<dim3(HIDDEN / 32, DIM / 32, NEXP), blk, 0, stream>>>(gate, gb, DIM, HIDDEN);
  transconv<<<dim3(HIDDEN / 32, DIM / 32, NEXP), blk, 0, stream>>>(up,   ub, DIM, HIDDEN);
  transconv<<<dim3(DIM / 32, HIDDEN / 32, NEXP), blk, 0, stream>>>(down, db, HIDDEN, DIM);

  moe_gateup<<<dim3(mblocks, HIDDEN / 128, NEXP), blk, 0, stream>>>(xb, gb, ub, cnts, hb);
  moe_down  <<<dim3(mblocks, DIM / 128, NEXP),    blk, 0, stream>>>(hb, db, cnts, out);
}
